// Gate_90640989815285
// MI455X (gfx1250) — matrix operand order repaired and checked by reading, not compiled, not run on hardware
//
#include <hip/hip_runtime.h>
#include <hip/hip_bf16.h>

#define D_DIM 7168
#define E_DIM 256

typedef __attribute__((ext_vector_type(16))) __bf16 v16bf;
typedef __attribute__((ext_vector_type(8)))  float  v8f;

union Frag { uint4 q[2]; v16bf v; };

// pack two f32 into one dword of two bf16 (truncation) with a single v_perm_b32
__device__ __forceinline__ unsigned pk_bf16(float lo, float hi) {
  return __builtin_amdgcn_perm(__float_as_uint(hi), __float_as_uint(lo), 0x07060302u);
}

__device__ __forceinline__ v8f wmma_bf16(v16bf a, v16bf b, v8f c) {
  // (neg_a, A, neg_b, B, c_mod, C, reuse_a, reuse_b)
  return __builtin_amdgcn_wmma_f32_16x16x32_bf16(false, a, false, b, (short)0, c, false, false);
}

// ---------------------------------------------------------------------------
// Split-K GEMM: partial[s][B,256] = x[B, ks:ke] * weight[256, ks:ke]^T
// grid = (B/128, S), 128 threads (4 waves). Block tile: 128 x 256, K-step 32.
// Wave w: M-tiles [ (w>>1)*4, +4 ), N-tiles [ (w&1)*8, +8 )  -> 32 WMMA/step.
// Software pipelined: global->regs for step k+1 overlaps WMMA on step k,
// LDS double buffered, one barrier per K-step.
// ---------------------------------------------------------------------------
__global__ __launch_bounds__(128) void gemm_bf16_wmma(
    const float* __restrict__ x, const float* __restrict__ wgt,
    float* __restrict__ scores, int kLen, long sStride)
{
  __shared__ unsigned As[2][128 * 16];    // 128 rows x 32 bf16 per buffer
  __shared__ unsigned Bs[2][E_DIM * 16];  // 256 rows x 32 bf16 per buffer

  const int t    = threadIdx.x;
  const int lane = t & 31;
  const int half = lane >> 4;
  const int l16  = lane & 15;
  const int wv   = t >> 5;
  const int mg   = (wv >> 1) * 4;      // wave's first M-tile (owns 4)
  const int ng   = (wv & 1) * 8;       // wave's first N-tile (owns 8)
  const long rowBase = (long)blockIdx.x * 128;
  const int  kBegin  = blockIdx.y * kLen;
  const int  kEnd    = kBegin + kLen;
  float* out = scores + (long)blockIdx.y * sStride;

  v8f acc[4][8];
  #pragma unroll
  for (int i = 0; i < 4; ++i)
    #pragma unroll
    for (int j = 0; j < 8; ++j)
      #pragma unroll
      for (int e = 0; e < 8; ++e) acc[i][j][e] = 0.0f;

  // staging: thread t owns A row t (32 floats) and B rows 2t, 2t+1 (32 floats each)
  const float* ag = x + (rowBase + t) * (long)D_DIM;
  const float* bg = wgt + (long)(2 * t) * D_DIM;

  float4 ta[8], tb[16];

  auto load_temps = [&](int k0) {
    const float4* pa  = (const float4*)(ag + k0);
    #pragma unroll
    for (int i = 0; i < 8; ++i) ta[i] = pa[i];
    const float4* pb0 = (const float4*)(bg + k0);
    const float4* pb1 = (const float4*)(bg + (long)D_DIM + k0);
    #pragma unroll
    for (int i = 0; i < 8; ++i) tb[i]     = pb0[i];
    #pragma unroll
    for (int i = 0; i < 8; ++i) tb[8 + i] = pb1[i];
  };

  auto cvt_store = [&](int buf) {
    uint4* qa = (uint4*)(As[buf] + t * 16);
    #pragma unroll
    for (int i = 0; i < 4; ++i)
      qa[i] = make_uint4(pk_bf16(ta[2*i].x,   ta[2*i].y),
                         pk_bf16(ta[2*i].z,   ta[2*i].w),
                         pk_bf16(ta[2*i+1].x, ta[2*i+1].y),
                         pk_bf16(ta[2*i+1].z, ta[2*i+1].w));
    uint4* qb = (uint4*)(Bs[buf] + 2 * t * 16);
    #pragma unroll
    for (int i = 0; i < 8; ++i)
      qb[i] = make_uint4(pk_bf16(tb[2*i].x,   tb[2*i].y),
                         pk_bf16(tb[2*i].z,   tb[2*i].w),
                         pk_bf16(tb[2*i+1].x, tb[2*i+1].y),
                         pk_bf16(tb[2*i+1].z, tb[2*i+1].w));
  };

  auto compute = [&](int buf) {
    // A frag layout (16-bit A 16x32):
    // lanes 0-15: elems 0..7 = K0..7,  8..15 = K16..23
    // lanes16-31: elems 0..7 = K8..15, 8..15 = K24..31
    Frag a[4];
    #pragma unroll
    for (int mi = 0; mi < 4; ++mi) {
      const unsigned* p = As[buf] + ((mg + mi) * 16 + l16) * 16 + half * 4;
      a[mi].q[0] = *(const uint4*)(p);
      a[mi].q[1] = *(const uint4*)(p + 8);
    }
    #pragma unroll
    for (int nt = 0; nt < 8; ++nt) {
      // B frag (16-bit B 32x16): lanes 0-15 K0..7,16..23, lanes 16-31 K8..15,24..31, N=lane&15
      Frag b;
      const unsigned* p = Bs[buf] + ((ng + nt) * 16 + l16) * 16 + half * 4;
      b.q[0] = *(const uint4*)(p);
      b.q[1] = *(const uint4*)(p + 8);
      #pragma unroll
      for (int mi = 0; mi < 4; ++mi)
        acc[mi][nt] = wmma_bf16(a[mi].v, b.v, acc[mi][nt]);
    }
  };

  load_temps(kBegin);
  cvt_store(0);
  __syncthreads();

  int cur = 0;
  for (int k0 = kBegin + 32; k0 < kEnd; k0 += 32) {
    load_temps(k0);                       // next step's globals (overlaps WMMAs)
    if (k0 + 32 < kEnd) {                 // prefetch 2 steps ahead of the x stream
      __builtin_prefetch(ag + k0 + 32, 0, 3);
    }
    compute(cur);                         // WMMAs on current LDS buffer
    cvt_store(cur ^ 1);                   // fill the other buffer
    __syncthreads();
    cur ^= 1;
  }
  compute(cur);

  // epilogue: C layout VGPR j -> M=j (lanes<16) / M=j+8 (lanes>=16), N=lane&15
  #pragma unroll
  for (int mi = 0; mi < 4; ++mi)
    #pragma unroll
    for (int nt = 0; nt < 8; ++nt)
      #pragma unroll
      for (int j = 0; j < 8; ++j) {
        long r = rowBase + (mg + mi) * 16 + half * 8 + j;
        int  c = (ng + nt) * 16 + l16;
        out[r * E_DIM + c] = acc[mi][nt][j];
      }
}

// ---------------------------------------------------------------------------
// Routing: one wave32 per row. lane owns experts [lane*8, lane*8+8) (all in
// group lane>>2). Sums S split-K partials (fixed order -> deterministic), then
// grouped top-2 sums -> top-4 groups -> masked top-8 -> normalized weights.
// ---------------------------------------------------------------------------
__global__ __launch_bounds__(256) void moe_route(
    const float* __restrict__ scores, const float* __restrict__ bias,
    float* __restrict__ outw, int* __restrict__ outidx, int S, long sStride)
{
  const int lane = threadIdx.x & 31;
  const int wave = threadIdx.x >> 5;
  const long row = (long)blockIdx.x * 8 + wave;
  const float* sr = scores + row * E_DIM;
  const float NEG = -__builtin_inff();

  float s[8], sb[8];
  #pragma unroll
  for (int i = 0; i < 8; ++i) {
    float v = 0.0f;
    for (int p = 0; p < S; ++p) v += sr[(long)p * sStride + lane * 8 + i];
    float sg = 1.0f / (1.0f + __expf(-v));
    s[i]  = sg;
    sb[i] = sg + bias[lane * 8 + i];
  }

  // local top-2 of this lane's 8 biased scores
  float m1 = NEG, m2 = NEG;
  #pragma unroll
  for (int i = 0; i < 8; ++i) {
    if (sb[i] > m1)      { m2 = m1; m1 = sb[i]; }
    else if (sb[i] > m2) { m2 = sb[i]; }
  }
  // merge top-2 across the 4 lanes of each group (xor 1, 2 stay in-group)
  #pragma unroll
  for (int off = 1; off <= 2; off <<= 1) {
    float o1 = __shfl_xor(m1, off, 32);
    float o2 = __shfl_xor(m2, off, 32);
    float hi = fmaxf(m1, o1), lo = fminf(m1, o1);
    m1 = hi;
    m2 = fmaxf(lo, fmaxf(m2, o2));
  }
  float gscore = m1 + m2;
  const int myg = lane >> 2;

  float gs[8];
  #pragma unroll
  for (int g = 0; g < 8; ++g) gs[g] = __shfl(gscore, g * 4, 32);
  int rank = 0;
  #pragma unroll
  for (int g = 0; g < 8; ++g)
    if (g != myg && (gs[g] > gs[myg] || (gs[g] == gs[myg] && g < myg))) rank++;
  const bool sel = rank < 4;

  // iterative masked top-8 over biased scores; carry sigmoid value alongside
  unsigned dead = 0;
  float myw = 0.0f, wsum = 0.0f;
  int myi = 0;
  #pragma unroll
  for (int k = 0; k < 8; ++k) {
    float bv = NEG, bs = 0.0f; int bi = 0;
    #pragma unroll
    for (int i = 0; i < 8; ++i) {
      float c = (sel && !((dead >> i) & 1u)) ? sb[i] : NEG;
      if (c > bv) { bv = c; bi = lane * 8 + i; bs = s[i]; }
    }
    #pragma unroll
    for (int off = 16; off; off >>= 1) {
      float ov = __shfl_xor(bv, off, 32);
      int   oi = __shfl_xor(bi, off, 32);
      float os = __shfl_xor(bs, off, 32);
      if (ov > bv || (ov == bv && oi < bi)) { bv = ov; bi = oi; bs = os; }
    }
    wsum += bs;
    if (lane == k) { myw = bs; myi = bi; }
    if ((bi >> 3) == lane) dead |= 1u << (bi & 7);
  }

  if (lane < 8) {
    outw[row * 8 + lane]   = myw * 2.5f / wsum;
    outidx[row * 8 + lane] = myi;
  }
}

extern "C" void kernel_launch(void* const* d_in, const int* in_sizes, int n_in,
                              void* d_out, int out_size, void* d_ws, size_t ws_size,
                              hipStream_t stream) {
  const float* x    = (const float*)d_in[0];
  const float* wgt  = (const float*)d_in[1];
  const float* bias = (const float*)d_in[2];
  const int B = in_sizes[0] / D_DIM;   // 8192

  const size_t per = (size_t)B * E_DIM * sizeof(float);  // one partial: 8 MB
  int S = (ws_size >= 4 * per) ? 4 : (ws_size >= 2 * per) ? 2 : 1;

  float* scores = (float*)d_ws;        // S x B x 256 f32 partials
  float* outw   = (float*)d_out;                    // B x 8 weights (f32)
  int*   outidx = (int*)(outw + (size_t)B * 8);     // B x 8 indices (i32 bits)

  const long sStride = (long)B * E_DIM;
  dim3 grid(B / 128, S);
  gemm_bf16_wmma<<<grid, 128, 0, stream>>>(x, wgt, scores, D_DIM / S, sStride);
  moe_route<<<B / 8, 256, 0, stream>>>(scores, bias, outw, outidx, S, sStride);
}
